// GPTUniAttention_6227702579436
// MI455X (gfx1250) — matrix operand order repaired and checked by reading, not compiled, not run on hardware
//
#include <hip/hip_runtime.h>
#include <cstdint>
#include <cstddef>

// ---------------------------------------------------------------------------
// GPT windowed-attention block for MI455X (gfx1250, wave32, WMMA bf16).
//   v   = X @ Wfc^T + bfc                (bf16 WMMA, f32 accum)
//   att = uniform-causal-window mean(v)  (prefix-sum trick, exact)
//   out = att @ Wproj^T + bproj          (bf16 WMMA, f32 accum)
//   also emit dense attn_weights [1,H,S,S]
// GEMMs use a double-buffered GLOBAL_LOAD_ASYNC_TO_LDS_B128 pipeline
// (ASYNCcnt) so HBM fetch of tile k+1 overlaps the 8 WMMAs of tile k.
// ---------------------------------------------------------------------------

typedef __attribute__((ext_vector_type(16))) __bf16       v16bf;
typedef __attribute__((ext_vector_type(8)))  float        v8f;
typedef __attribute__((ext_vector_type(4)))  unsigned int u32x4;
typedef __attribute__((ext_vector_type(4)))  float        f32x4;
typedef __attribute__((ext_vector_type(4)))  __bf16       bf16x4;

#define BATCH   2
#define SEQ     2048
#define DIM     1024
#define HEADS   16
#define MROWS   (BATCH * SEQ)   // 4096
#define CH      256             // scan chunk length
#define NCH     (SEQ / CH)      // 8

// ---------------- fp32 -> bf16 pre-conversion (vectorized) -----------------
__global__ void k_f32_to_bf16(const float* __restrict__ in,
                              __bf16* __restrict__ out, int n4) {
  int i = blockIdx.x * blockDim.x + threadIdx.x;
  if (i >= n4) return;
  f32x4 f = ((const f32x4*)in)[i];
  bf16x4 o;
  o.x = (__bf16)f.x; o.y = (__bf16)f.y; o.z = (__bf16)f.z; o.w = (__bf16)f.w;
  ((bf16x4*)out)[i] = o;
}

// ---------------- bf16 WMMA GEMM:  C[M,N] = A[M,K] @ W[N,K]^T + bias -------
#define BM 128
#define BN 128
#define BK 32
#define LDSS 48   // halfwords per LDS row (96B: 16B-aligned, conflict-padded)

union FragU { u32x4 q[2]; v16bf v; };

// LDS byte offset of a __shared__ pointer (generic -> addrspace(3) -> int)
__device__ __forceinline__ unsigned lds_off(const void* p) {
  return (unsigned)(unsigned long long)(const __attribute__((address_space(3))) void*)p;
}

// Issue the async LDS fill for one (A,W) K-tile: 4 x b128 per thread.
// IOFFSET applies to both LDS and global addresses, so each pair of 16B
// transfers shares one LDS-addr VGPR and one global-addr VGPR pair.
__device__ __forceinline__ void async_tile_load(unsigned ldsA, unsigned ldsW,
                                                const __bf16* ag,
                                                const __bf16* wg) {
  asm volatile(
      "global_load_async_to_lds_b128 %0, %2, off\n\t"
      "global_load_async_to_lds_b128 %0, %2, off offset:16\n\t"
      "global_load_async_to_lds_b128 %1, %3, off\n\t"
      "global_load_async_to_lds_b128 %1, %3, off offset:16"
      :
      : "v"(ldsA), "v"(ldsW), "v"(ag), "v"(wg)
      : "memory");
}

__global__ __launch_bounds__(256)
void k_gemm_bf16(const __bf16* __restrict__ A, const __bf16* __restrict__ W,
                 const float* __restrict__ bias, float* __restrict__ C,
                 int M, int N, int K) {
  __shared__ __attribute__((aligned(16))) __bf16 As[2][BM * LDSS];
  __shared__ __attribute__((aligned(16))) __bf16 Ws[2][BN * LDSS];

  const int tid  = threadIdx.x;
  const int lane = tid & 31;
  const int wave = tid >> 5;      // 0..7
  const int wm   = wave >> 2;     // 0..1  -> 64-row slab
  const int wn   = wave & 3;      // 0..3  -> 32-col slab
  const int half = lane >> 4;     // lane half selects K group (ISA 7.12.2)
  const int l15  = lane & 15;

  const int blockM = blockIdx.y * BM;
  const int blockN = blockIdx.x * BN;

  const int lrow = tid >> 1;        // 0..127: tile row loaded by this thread
  const int lseg = (tid & 1) * 16;  // 16-halfword segment within BK

  // per-thread LDS destinations (both buffers) and global row pointers
  const unsigned ldsA0 = lds_off(&As[0][lrow * LDSS + lseg]);
  const unsigned ldsA1 = lds_off(&As[1][lrow * LDSS + lseg]);
  const unsigned ldsW0 = lds_off(&Ws[0][lrow * LDSS + lseg]);
  const unsigned ldsW1 = lds_off(&Ws[1][lrow * LDSS + lseg]);
  const __bf16* ag = A + (size_t)(blockM + lrow) * K + lseg;
  const __bf16* wg = W + (size_t)(blockN + lrow) * K + lseg;

  v8f zero = {0.f,0.f,0.f,0.f,0.f,0.f,0.f,0.f};
  v8f acc[4][2];
  #pragma unroll
  for (int mi = 0; mi < 4; ++mi)
    #pragma unroll
    for (int ni = 0; ni < 2; ++ni) acc[mi][ni] = zero;

  const int KT = K / BK;
  async_tile_load(ldsA0, ldsW0, ag, wg);          // prologue: tile 0 -> buf 0

  for (int kt = 0; kt < KT; ++kt) {
    // my tile-kt fill is the only outstanding async work
    asm volatile("s_wait_asynccnt 0x0" ::: "memory");
    // all waves' tile-kt LDS writes visible; also gates overwrite of buf (kt+1)&1
    __syncthreads();

    if (kt + 1 < KT) {                            // fill other buffer async
      const unsigned la = (kt & 1) ? ldsA0 : ldsA1;
      const unsigned lw = (kt & 1) ? ldsW0 : ldsW1;
      async_tile_load(la, lw, ag + (kt + 1) * BK, wg + (kt + 1) * BK);
    }

    const __bf16* Ab = As[kt & 1];
    const __bf16* Wb = Ws[kt & 1];

    // A fragment: 16x32, lane half -> K base {0,8}; e: K = kb + (e>>3)*16 + (e&7)
    v16bf afrag[4];
    #pragma unroll
    for (int mi = 0; mi < 4; ++mi) {
      const int row = wm * 64 + mi * 16 + l15;
      const int kb  = half * 8;
      FragU u;
      u.q[0] = *(const u32x4*)&Ab[row * LDSS + kb];
      u.q[1] = *(const u32x4*)&Ab[row * LDSS + kb + 16];
      afrag[mi] = u.v;
    }
    // B fragment: 32x16 (K-major from W rows); lane half -> K base {0,8}; e: K = kb + (e>>3)*16 + (e&7)
    v16bf bfrag[2];
    #pragma unroll
    for (int ni = 0; ni < 2; ++ni) {
      const int col = wn * 32 + ni * 16 + l15;
      const int kb  = half * 8;
      FragU u;
      u.q[0] = *(const u32x4*)&Wb[col * LDSS + kb];
      u.q[1] = *(const u32x4*)&Wb[col * LDSS + kb + 16];
      bfrag[ni] = u.v;
    }
    #pragma unroll
    for (int mi = 0; mi < 4; ++mi)
      #pragma unroll
      for (int ni = 0; ni < 2; ++ni)
        acc[mi][ni] = __builtin_amdgcn_wmma_f32_16x16x32_bf16(
            false, afrag[mi], false, bfrag[ni], (short)0, acc[mi][ni],
            false, false);
  }

  // C/D layout: VGPR r -> M = half*8 + r, N = l15
  #pragma unroll
  for (int mi = 0; mi < 4; ++mi) {
    #pragma unroll
    for (int ni = 0; ni < 2; ++ni) {
      const int row0 = blockM + wm * 64 + mi * 16 + half * 8;
      const int col  = blockN + wn * 32 + ni * 16 + l15;
      const float bv = bias[col];
      #pragma unroll
      for (int r = 0; r < 8; ++r)
        C[(size_t)(row0 + r) * N + col] = acc[mi][ni][r] + bv;
    }
  }
}

// ---------------- windowed attention via prefix sums -----------------------
// pass 1: per-(b,chunk,n) partial sums (coalesced along n)
__global__ void k_scan_chunks(const float* __restrict__ v,
                              float* __restrict__ csum) {
  int idx = blockIdx.x * 256 + threadIdx.x;   // (b*NCH + c)*DIM + n
  int n = idx & (DIM - 1);
  int t = idx >> 10;
  int c = t & (NCH - 1);
  int b = t >> 3;
  if (b >= BATCH) return;
  const float* p = v + ((size_t)(b * SEQ + c * CH)) * DIM + n;
  float s = 0.f;
  for (int i = 0; i < CH; ++i) s += p[(size_t)i * DIM];
  csum[idx] = s;
}

// pass 2: inclusive prefix P[b,s,n] over the sequence
__global__ void k_scan_finalize(const float* __restrict__ v,
                                const float* __restrict__ csum,
                                float* __restrict__ P) {
  int idx = blockIdx.x * 256 + threadIdx.x;
  int n = idx & (DIM - 1);
  int t = idx >> 10;
  int c = t & (NCH - 1);
  int b = t >> 3;
  if (b >= BATCH) return;
  float run = 0.f;
  for (int cc = 0; cc < c; ++cc) run += csum[(b * NCH + cc) * DIM + n];
  const float* src = v + ((size_t)(b * SEQ + c * CH)) * DIM + n;
  float*       dst = P + ((size_t)(b * SEQ + c * CH)) * DIM + n;
  for (int i = 0; i < CH; ++i) {
    run += src[(size_t)i * DIM];
    dst[(size_t)i * DIM] = run;
  }
}

// pass 3: att[b,s,n] = (P[s] - P[s-L]) / L,  L = min(s+1, 2^(h+1)-1); emit bf16
__global__ void k_attn_window(const float* __restrict__ P,
                              __bf16* __restrict__ AT) {
  size_t idx = (size_t)blockIdx.x * 256 + threadIdx.x;  // b*S*D + s*D + n
  int n = (int)(idx & (DIM - 1));
  int s = (int)((idx >> 10) & (SEQ - 1));
  int h = n >> 6;                       // 64 = head_dim
  int wsz = (2 << h) - 1;               // 2^(h+1)-1
  int L = (s + 1 < wsz) ? s + 1 : wsz;
  float upper = P[idx];
  float lower = (L <= s) ? P[idx - (size_t)L * DIM] : 0.f;
  AT[idx] = (__bf16)((upper - lower) / (float)L);
}

// ---------------- dense attn_weights output [1,H,S,S] ----------------------
__global__ void k_fill_weights(float* __restrict__ wout) {
  size_t t = (size_t)blockIdx.x * 256 + threadIdx.x;
  size_t base = t * 4;                       // 4 floats per thread
  int j0 = (int)(base & (SEQ - 1));
  size_t r = base >> 11;                     // h*S + i
  int i = (int)(r & (SEQ - 1));
  int h = (int)(r >> 11);
  int wsz = (2 << h) - 1;
  int vl = (i + 1 < wsz) ? i + 1 : wsz;
  float inv = 1.0f / (float)vl;
  f32x4 o;
  #pragma unroll
  for (int q = 0; q < 4; ++q) {
    int d = i - (j0 + q);
    o[q] = (d >= 0 && d < wsz) ? inv : 0.f;
  }
  *(f32x4*)&wout[base] = o;
}

// ---------------------------------------------------------------------------
extern "C" void kernel_launch(void* const* d_in, const int* in_sizes, int n_in,
                              void* d_out, int out_size, void* d_ws,
                              size_t ws_size, hipStream_t stream) {
  const float* X     = (const float*)d_in[0];  // [B,S,D]
  const float* Wfc   = (const float*)d_in[1];  // [D,D]
  const float* bfc   = (const float*)d_in[2];  // [D]
  const float* Wproj = (const float*)d_in[3];  // [D,D]
  const float* bproj = (const float*)d_in[4];  // [D]

  float* out  = (float*)d_out;
  float* wout = out + (size_t)MROWS * DIM;     // attn_weights segment

  char* ws = (char*)d_ws;
  __bf16* Xbf   = (__bf16*)(ws);                          //  0 .. 8 MiB
  __bf16* Wfcb  = (__bf16*)(ws + ((size_t)8  << 20));     //  8 .. 10 MiB
  __bf16* Wprb  = (__bf16*)(ws + ((size_t)10 << 20));     // 10 .. 12 MiB
  float*  v     = (float*) (ws + ((size_t)12 << 20));     // 12 .. 28 MiB
  float*  P     = (float*) (ws + ((size_t)28 << 20));     // 28 .. 44 MiB
  float*  csum  = (float*) (ws + ((size_t)44 << 20));     // 64 KiB
  __bf16* AT    = (__bf16*)(ws + ((size_t)45 << 20));     // 45 .. 53 MiB

  // 1) one-time bf16 conversions
  k_f32_to_bf16<<<(MROWS * DIM / 4) / 256, 256, 0, stream>>>(X, Xbf,
                                                             MROWS * DIM / 4);
  k_f32_to_bf16<<<(DIM * DIM / 4) / 256, 256, 0, stream>>>(Wfc, Wfcb,
                                                           DIM * DIM / 4);
  k_f32_to_bf16<<<(DIM * DIM / 4) / 256, 256, 0, stream>>>(Wproj, Wprb,
                                                           DIM * DIM / 4);
  // 2) v = X @ Wfc^T + bfc
  dim3 gg(DIM / BN, MROWS / BM);
  k_gemm_bf16<<<gg, 256, 0, stream>>>(Xbf, Wfcb, bfc, v, MROWS, DIM, DIM);

  // 3) windowed mean via prefix sums
  k_scan_chunks  <<<(BATCH * NCH * DIM) / 256, 256, 0, stream>>>(v, csum);
  k_scan_finalize<<<(BATCH * NCH * DIM) / 256, 256, 0, stream>>>(v, csum, P);
  k_attn_window  <<<(MROWS * DIM) / 256, 256, 0, stream>>>(P, AT);

  // 4) out = att @ Wproj^T + bproj
  k_gemm_bf16<<<gg, 256, 0, stream>>>(AT, Wprb, bproj, out, MROWS, DIM, DIM);

  // 5) dense attention-weight tensor
  k_fill_weights<<<((size_t)HEADS * SEQ * SEQ / 4) / 256, 256, 0, stream>>>(
      wout);
}
